// TMnet_90984587199099
// MI455X (gfx1250) — matrix operand order repaired and checked by reading, not compiled, not run on hardware
//
#include <hip/hip_runtime.h>
#include <math.h>

// ---------------------------------------------------------------------------
// Shapes (fixed by the reference)
// ---------------------------------------------------------------------------
#define DIMC   384
#define NHEAD  8
#define HDIM   48          // DIM / NH
#define HH     128
#define NPX    16384       // H*W
#define HIDC   1152        // 3*DIM

typedef __attribute__((ext_vector_type(16))) _Float16 v16h;
typedef __attribute__((ext_vector_type(8)))  float    v8f;
typedef __attribute__((ext_vector_type(2)))  float    v2f;
typedef __attribute__((ext_vector_type(4)))  unsigned int u32x4;
typedef __attribute__((ext_vector_type(8)))  int      i32x8;
typedef __attribute__((ext_vector_type(4)))  int      i32x4;

__device__ __forceinline__ float gelu_exact(float v) {
    return 0.5f * v * (1.0f + erff(v * 0.70710678118654752f));
}

// ---------------------------------------------------------------------------
// f32 -> f16 convert (weights, etc.)
// ---------------------------------------------------------------------------
__global__ void k_cvt_f16(const float* __restrict__ in, _Float16* __restrict__ out, size_t n) {
    size_t i = (size_t)blockIdx.x * 256 + threadIdx.x;
    if (i < n) out[i] = (_Float16)in[i];
}

// ---------------------------------------------------------------------------
// LayerNorm over channels (per pixel), writes f16 for the following GEMM
// ---------------------------------------------------------------------------
__global__ void k_layernorm(const float* __restrict__ x, const float* __restrict__ w,
                            const float* __restrict__ b, _Float16* __restrict__ y,
                            int C, int N) {
    int n = blockIdx.x * 256 + threadIdx.x;
    if (n >= N) return;
    float s = 0.f;
    for (int c = 0; c < C; ++c) s += x[(size_t)c * N + n];
    float mu = s / (float)C;
    float v = 0.f;
    for (int c = 0; c < C; ++c) { float d = x[(size_t)c * N + n] - mu; v += d * d; }
    float inv = rsqrtf(v / (float)C + 1e-5f);
    for (int c = 0; c < C; ++c) {
        float t = (x[(size_t)c * N + n] - mu) * inv * w[c] + b[c];
        y[(size_t)c * N + n] = (_Float16)t;
    }
}

// ---------------------------------------------------------------------------
// WMMA f16 GEMM:  out[M,N] = W[M,K] * X[K,N] + bias[M]  (+ optional residual)
// 256 threads = 8 waves; workgroup tile 64(M) x 64(N); wave tile 16x32.
//
// A tiles (64x32 f16, row stride K) are staged into LDS by the Tensor Data
// Mover: one tensor_load_to_lds per K-chunk, double-buffered and pipelined
// with s_wait_tensorcnt (chunk i+1 issued before waiting on chunk i).
// TDM pad fields insert 16B per 64B row -> LDS row stride 40 f16 (20 DWORDs,
// conflict-free).  B tiles are transposed through VGPRs into [n][k] LDS so
// each B fragment is two aligned 16B ds reads.
// ---------------------------------------------------------------------------
__global__ void k_gemm_f16(const _Float16* __restrict__ W, const _Float16* __restrict__ X,
                           const float* __restrict__ bias, const float* __restrict__ resid,
                           float* __restrict__ out, int M, int K, int N) {
    __shared__ _Float16 As [2][64 * 40];   // [m][k] stride 40 (TDM-padded)
    __shared__ _Float16 BsT[2][64 * 40];   // [n][k] stride 40

    const int tid  = threadIdx.x;
    const int lane = tid & 31;
    const int wave = tid >> 5;
    const int wm   = wave >> 1;         // 0..3  (M sub-tile)
    const int wn   = wave & 1;          // 0..1  (N sub-tile pair)
    const int hf   = lane >> 4;         // half-wave
    const int l16  = lane & 15;

    const int m0 = blockIdx.y * 64;
    const int n0 = blockIdx.x * 64;

    const int brow  = tid >> 3;          // 32 rows (k) of B
    const int bcol8 = (tid & 7) * 8;     // 64 cols (n), 8 f16 each

    // ---- TDM descriptor issue: A tile [64 x 32] f16 at (m0, k0) -> As[buf]
    auto issue_tdm_a = [&](int k0, int buf) {
        const unsigned lds = (unsigned)(uintptr_t)&As[buf][0];
        const unsigned long long ga =
            (unsigned long long)(uintptr_t)(W + (size_t)m0 * K + k0);
        u32x4 g0;
        g0.x = 1u;                                        // count = 1 (valid D#)
        g0.y = lds;                                       // lds_addr (bytes)
        g0.z = (unsigned)(ga & 0xffffffffu);              // global_addr[31:0]
        g0.w = (unsigned)((ga >> 32) & 0x01ffffffu)       // global_addr[56:32]
             | (2u << 30);                                // type = 2 (image)
        i32x8 g1;
        g1[0] = (1 << 16)        // data_size = 2 bytes
              | (1 << 20)        // pad_enable
              | (3 << 22)        // pad_interval: 16 DWORDs (one 64B row)
              | (3 << 25);       // pad_amount: 4 DWORDs (16B) -> stride 40 f16
        g1[1] = (int)(((unsigned)K & 0xffffu) << 16);     // tensor_dim0 lo16
        g1[2] = (int)((((unsigned)K >> 16) & 0xffffu)     // tensor_dim0 hi16
              |       (((unsigned)M & 0xffffu) << 16));   // tensor_dim1 lo16
        g1[3] = (int)((((unsigned)M >> 16) & 0xffffu)     // tensor_dim1 hi16
              |       (32u << 16));                       // tile_dim0 = 32 elems
        g1[4] = 64;                                       // tile_dim1 = 64 rows
        g1[5] = K;                                        // tensor_dim0_stride lo32
        g1[6] = 0;
        g1[7] = 0;
        const i32x4 z4 = {0, 0, 0, 0};
#if defined(__clang_major__) && __clang_major__ >= 23
        const i32x8 z8 = {0, 0, 0, 0, 0, 0, 0, 0};
        __builtin_amdgcn_tensor_load_to_lds(g0, g1, z4, z4, z8, 0);
#else
        __builtin_amdgcn_tensor_load_to_lds(g0, g1, z4, z4, 0);
#endif
    };

    v8f acc0 = {}; v8f acc1 = {};

    if (tid == 0) issue_tdm_a(0, 0);                      // prologue

    for (int k0 = 0, it = 0; k0 < K; k0 += 32, ++it) {
        const int buf = it & 1;

        // ---- stage B tile (transpose into [n][k]) — all threads
        union { float4 f; _Float16 h[8]; } xb;
        xb.f = *(const float4*)(X + (size_t)(k0 + brow) * N + n0 + bcol8);
        #pragma unroll
        for (int e = 0; e < 8; ++e) BsT[buf][(bcol8 + e) * 40 + brow] = xb.h[e];
        if (k0 + 32 < K)
            __builtin_prefetch(X + (size_t)(k0 + 32 + brow) * N + n0 + bcol8, 0, 1);

        // ---- pipeline the next A tile, then wait for the current one
        if (tid == 0) {
            if (k0 + 32 < K) {
                issue_tdm_a(k0 + 32, buf ^ 1);
                __builtin_amdgcn_s_wait_tensorcnt(1);     // chunk `it` complete
            } else {
                __builtin_amdgcn_s_wait_tensorcnt(0);
            }
        }
        __syncthreads();

        // ---- fragments
        union { v16h v; float4 f[2]; } af, b0, b1;
        const int ar = (wm * 16 + l16) * 40;
        af.f[0] = *(const float4*)&As[buf][ar + 8 * hf];        // K = 8h..8h+7
        af.f[1] = *(const float4*)&As[buf][ar + 16 + 8 * hf];   // K = 16+8h..+7
        const int nc0 = (wn * 32 + l16) * 40;
        const int nc1 = (wn * 32 + 16 + l16) * 40;
        b0.f[0] = *(const float4*)&BsT[buf][nc0 + 8 * hf];      // K = 8h..8h+7
        b0.f[1] = *(const float4*)&BsT[buf][nc0 + 16 + 8 * hf]; // K = 16+8h..+7
        b1.f[0] = *(const float4*)&BsT[buf][nc1 + 8 * hf];
        b1.f[1] = *(const float4*)&BsT[buf][nc1 + 16 + 8 * hf];

        acc0 = __builtin_amdgcn_wmma_f32_16x16x32_f16(false, af.v, false, b0.v,
                                                      (short)0, acc0, false, false);
        acc1 = __builtin_amdgcn_wmma_f32_16x16x32_f16(false, af.v, false, b1.v,
                                                      (short)0, acc1, false, false);
        __syncthreads();
    }

    // ---- epilogue: C/D layout M = r + 8*hf, N = l16
    #pragma unroll
    for (int r = 0; r < 8; ++r) {
        const int m = m0 + wm * 16 + hf * 8 + r;
        const size_t o0 = (size_t)m * N + n0 + wn * 32 + l16;
        float v0 = acc0[r] + bias[m];
        float v1 = acc1[r] + bias[m];
        if (resid) { v0 += resid[o0]; v1 += resid[o0 + 16]; }
        out[o0]      = v0;
        out[o0 + 16] = v1;
    }
}

// ---------------------------------------------------------------------------
// Depthwise 3x3 SAME conv (optionally over a concat of two inputs)
// ---------------------------------------------------------------------------
__global__ void k_dw3x3(const float* __restrict__ in0, const float* __restrict__ in1,
                        int split, const float* __restrict__ w, const float* __restrict__ b,
                        float* __restrict__ out, int C) {
    size_t idx = (size_t)blockIdx.x * 256 + threadIdx.x;
    if (idx >= (size_t)C * NPX) return;
    int c = (int)(idx >> 14);
    int p = (int)(idx & 16383);
    int y = p >> 7, x = p & 127;
    const float* src = (c < split) ? (in0 + (size_t)c * NPX)
                                   : (in1 + (size_t)(c - split) * NPX);
    const float* wc = w + (size_t)c * 9;
    float s = b[c];
    #pragma unroll
    for (int ky = 0; ky < 3; ++ky) {
        int iy = y + ky - 1;
        if ((unsigned)iy >= (unsigned)HH) continue;
        #pragma unroll
        for (int kx = 0; kx < 3; ++kx) {
            int ix = x + kx - 1;
            if ((unsigned)ix >= (unsigned)HH) continue;
            s += src[iy * HH + ix] * wc[ky * 3 + kx];
        }
    }
    out[idx] = s;
}

// ---------------------------------------------------------------------------
// GELU (exact) + cast to f16
// ---------------------------------------------------------------------------
__global__ void k_gelu_cast(const float* __restrict__ in, _Float16* __restrict__ out, size_t n) {
    size_t i = (size_t)blockIdx.x * 256 + threadIdx.x;
    if (i < n) out[i] = (_Float16)gelu_exact(in[i]);
}

// ---------------------------------------------------------------------------
// FFN gate: out = f16( gelu(h[:HID]) * h[HID:] )
// ---------------------------------------------------------------------------
__global__ void k_gate(const float* __restrict__ hg, _Float16* __restrict__ out,
                       size_t n, size_t off) {
    size_t i = (size_t)blockIdx.x * 256 + threadIdx.x;
    if (i < n) out[i] = (_Float16)(gelu_exact(hg[i]) * hg[i + off]);
}

// ---------------------------------------------------------------------------
// In-place L2 row normalization (row length NPX), one block per row
// ---------------------------------------------------------------------------
__global__ void k_rownorm(float* __restrict__ q, int N) {
    __shared__ float red[256];
    float* p = q + (size_t)blockIdx.x * N;
    float s = 0.f;
    for (int i = threadIdx.x; i < N; i += 256) { float t = p[i]; s += t * t; }
    red[threadIdx.x] = s;
    __syncthreads();
    for (int st = 128; st > 0; st >>= 1) {
        if (threadIdx.x < st) red[threadIdx.x] += red[threadIdx.x + st];
        __syncthreads();
    }
    float inv = 1.0f / fmaxf(sqrtf(red[0]), 1e-12f);
    for (int i = threadIdx.x; i < N; i += 256) p[i] *= inv;
}

// ---------------------------------------------------------------------------
// Channel attention logits: P[h,c,d] = temp[h] * <q_row(h,c), k_row(h,d)>
// ---------------------------------------------------------------------------
__global__ void k_attmat(const float* __restrict__ q, const float* __restrict__ k,
                         const float* __restrict__ temp, float* __restrict__ P, int N) {
    __shared__ float red[256];
    int idx = blockIdx.x;
    int h = idx / (HDIM * HDIM);
    int rem = idx % (HDIM * HDIM);
    int c = rem / HDIM, d = rem % HDIM;
    const float* qp = q + (size_t)(h * HDIM + c) * N;
    const float* kp = k + (size_t)(h * HDIM + d) * N;
    float s = 0.f;
    for (int i = threadIdx.x; i < N; i += 256) s += qp[i] * kp[i];
    red[threadIdx.x] = s;
    __syncthreads();
    for (int st = 128; st > 0; st >>= 1) {
        if (threadIdx.x < st) red[threadIdx.x] += red[threadIdx.x + st];
        __syncthreads();
    }
    if (threadIdx.x == 0) P[idx] = red[0] * temp[h];
}

// ---------------------------------------------------------------------------
// Per-row top-k mask + softmax (rows of length 48), one thread per row
// ---------------------------------------------------------------------------
__global__ void k_topk_softmax(float* __restrict__ P, const int* __restrict__ kptr, int rows) {
    int r = blockIdx.x * 128 + threadIdx.x;
    if (r >= rows) return;
    const int kk = *kptr;
    float* row = P + (size_t)r * HDIM;
    float a[HDIM];
    bool keep[HDIM];
    for (int i = 0; i < HDIM; ++i) a[i] = row[i];
    float mx = -3.4e38f;
    for (int i = 0; i < HDIM; ++i) {
        int cnt = 0;
        for (int j = 0; j < HDIM; ++j)
            if (a[j] > a[i] || (a[j] == a[i] && j < i)) ++cnt;
        keep[i] = (cnt < kk);
        if (keep[i]) mx = fmaxf(mx, a[i]);
    }
    float ssum = 0.f;
    for (int i = 0; i < HDIM; ++i) {
        if (keep[i]) { a[i] = expf(a[i] - mx); ssum += a[i]; }
    }
    float inv = 1.0f / ssum;
    for (int i = 0; i < HDIM; ++i) row[i] = keep[i] ? a[i] * inv : 0.f;
}

// ---------------------------------------------------------------------------
// attn @ V with f32 WMMA (16x16x4): out[h,c,n] = sum_d P[h,c,d] V[h,d,n]
// ---------------------------------------------------------------------------
__global__ void k_attn_av(const float* __restrict__ P, const float* __restrict__ V,
                          _Float16* __restrict__ out, int N) {
    const int lane = threadIdx.x & 31;
    const int wave = threadIdx.x >> 5;
    const int hf   = lane >> 4;
    const int l16  = lane & 15;
    const int n0 = blockIdx.x * 128 + wave * 16;
    const int c0 = blockIdx.y * 16;
    const int h  = blockIdx.z;
    const float* Ph = P + (size_t)h * HDIM * HDIM;
    const float* Vh = V + (size_t)h * HDIM * N;

    v8f acc = {};
    #pragma unroll
    for (int k0 = 0; k0 < HDIM; k0 += 4) {
        const int ka = k0 + 2 * hf;
        v2f a, b;
        a.x = Ph[(c0 + l16) * HDIM + ka];
        a.y = Ph[(c0 + l16) * HDIM + ka + 1];
        b.x = Vh[(size_t)ka * N + n0 + l16];
        b.y = Vh[(size_t)(ka + 1) * N + n0 + l16];
        acc = __builtin_amdgcn_wmma_f32_16x16x4_f32(false, a, false, b,
                                                    (short)0, acc, false, false);
    }
    #pragma unroll
    for (int r = 0; r < 8; ++r) {
        const int c = c0 + hf * 8 + r;
        out[(size_t)(h * HDIM + c) * N + n0 + l16] = (_Float16)acc[r];
    }
}

// ---------------------------------------------------------------------------
// Host-side orchestration
// ---------------------------------------------------------------------------
extern "C" void kernel_launch(void* const* d_in, const int* in_sizes, int n_in,
                              void* d_out, int out_size, void* d_ws, size_t ws_size,
                              hipStream_t stream) {
    const float* x  = (const float*)d_in[0];
    const float* f1 = (const float*)d_in[1];
    const float* f2 = (const float*)d_in[2];
    const float* f3 = (const float*)d_in[3];
    auto pp = [&](int i) { return (const float*)d_in[4 + i]; };
    const int* kkptr = (const int*)d_in[44];
    float* outp = (float*)d_out;

    // ---- workspace carve-up (bytes) -------------------------------------
    char* ws = (char*)d_ws;
    _Float16* WH = (_Float16*)ws;                          // f16 weight arena
    _Float16* yh = (_Float16*)(ws + 7372800);              // LN out [384,N] f16
    float*    G1 = (float*)   (ws + 19955712);             // 2304*N f32
    float*    G2 = (float*)   (ws + 170950656);            // 47,185,920 f32
    float*    Pm = (float*)   (ws + 359694336);            // logits [8,48,48]
    float*    X1 = (float*)   (ws + 359768064);            // residual [384,N]

    _Float16* aq_qkv_wh  = WH + 0;
    _Float16* akv_qkv_wh = WH + 442368;
    _Float16* aq_proj_wh = WH + 884736;
    _Float16* akv_proj_wh= WH + 1032192;
    _Float16* aq_q_pw_wh = WH + 1179648;
    _Float16* akv_k_pw_wh= WH + 1474560;
    _Float16* akv_v_pw_wh= WH + 1769472;
    _Float16* fus_pw_wh  = WH + 2064384;
    _Float16* ffn_in_wh  = WH + 2359296;
    _Float16* ffn_out_wh = WH + 3244032;

    float*    qkv   = G1;                                  // [1152,N]
    float*    qkvdw = G1 + (size_t)HIDC * NPX;             // [1152,N]
    float*    tdw   = G2;                                  // [768,N] f32
    _Float16* th    = (_Float16*)(G2 + 12582912);          // [768,N] f16
    float*    qn    = G2 + 18874368;                       // [384,N]
    float*    v2b   = G2 + 25165824;                       // [384,N]
    _Float16* attoh = (_Float16*)(G2 + 31457280);          // [384,N] f16
    float*    outq  = G2 + 34603008;                       // [384,N]
    float*    outkv = G2 + 40894464;                       // [384,N]
    float*    hgdw  = G2;                                  // stage D: [2304,N]
    _Float16* gh    = (_Float16*)(G2 + 37748736);          // stage D: [1152,N]

    const int N = NPX;
    auto cvt = [&](const float* s, _Float16* d, size_t n) {
        k_cvt_f16<<<(unsigned)((n + 255) / 256), 256, 0, stream>>>(s, d, n);
    };
    auto gemm = [&](const _Float16* W, const _Float16* X, const float* bias,
                    const float* resid, float* o, int M, int K) {
        k_gemm_f16<<<dim3(N / 64, M / 64), 256, 0, stream>>>(W, X, bias, resid, o, M, K, N);
    };
    auto dw = [&](const float* i0, const float* i1, int split, const float* w,
                  const float* b, float* o, int C) {
        k_dw3x3<<<(unsigned)(((size_t)C * NPX + 255) / 256), 256, 0, stream>>>(i0, i1, split, w, b, o, C);
    };
    auto gelu = [&](const float* i, _Float16* o, size_t n) {
        k_gelu_cast<<<(unsigned)((n + 255) / 256), 256, 0, stream>>>(i, o, n);
    };

    // ---- convert weights to f16 -----------------------------------------
    cvt(pp(0),  aq_qkv_wh,  (size_t)HIDC * DIMC);
    cvt(pp(7),  akv_qkv_wh, (size_t)HIDC * DIMC);
    cvt(pp(4),  aq_proj_wh, (size_t)DIMC * DIMC);
    cvt(pp(11), akv_proj_wh,(size_t)DIMC * DIMC);
    cvt(pp(16), aq_q_pw_wh, (size_t)DIMC * 2 * DIMC);
    cvt(pp(20), akv_k_pw_wh,(size_t)DIMC * 2 * DIMC);
    cvt(pp(24), akv_v_pw_wh,(size_t)DIMC * 2 * DIMC);
    cvt(pp(28), fus_pw_wh,  (size_t)DIMC * 2 * DIMC);
    cvt(pp(34), ffn_in_wh,  (size_t)2 * HIDC * DIMC);
    cvt(pp(38), ffn_out_wh, (size_t)DIMC * HIDC);

    // ---- norm1 ----------------------------------------------------------
    k_layernorm<<<N / 256, 256, 0, stream>>>(x, pp(30), pp(31), yh, DIMC, N);

    // ======================= aq branch ===================================
    gemm(aq_qkv_wh, yh, pp(1), nullptr, qkv, HIDC, DIMC);
    dw(qkv, nullptr, HIDC, pp(2), pp(3), qkvdw, HIDC);
    dw(qkvdw /*q*/, f1, DIMC, pp(14), pp(15), tdw, 2 * DIMC);
    gelu(tdw, th, (size_t)2 * DIMC * NPX);
    gemm(aq_q_pw_wh, th, pp(17), nullptr, qn, DIMC, 2 * DIMC);
    k_rownorm<<<DIMC, 256, 0, stream>>>(qn, N);
    k_rownorm<<<DIMC, 256, 0, stream>>>(qkvdw + (size_t)DIMC * NPX, N);
    k_attmat<<<NHEAD * HDIM * HDIM, 256, 0, stream>>>(qn, qkvdw + (size_t)DIMC * NPX, pp(6), Pm, N);
    k_topk_softmax<<<3, 128, 0, stream>>>(Pm, kkptr, NHEAD * HDIM);
    k_attn_av<<<dim3(N / 128, HDIM / 16, NHEAD), 256, 0, stream>>>(Pm, qkvdw + (size_t)2 * DIMC * NPX, attoh, N);
    gemm(aq_proj_wh, attoh, pp(5), nullptr, outq, DIMC, DIMC);

    // ======================= akv branch ==================================
    gemm(akv_qkv_wh, yh, pp(8), nullptr, qkv, HIDC, DIMC);
    dw(qkv, nullptr, HIDC, pp(9), pp(10), qkvdw, HIDC);
    dw(qkvdw + (size_t)DIMC * NPX /*k*/, f2, DIMC, pp(18), pp(19), tdw, 2 * DIMC);
    gelu(tdw, th, (size_t)2 * DIMC * NPX);
    gemm(akv_k_pw_wh, th, pp(21), nullptr, qn /*k2*/, DIMC, 2 * DIMC);
    dw(qkvdw + (size_t)2 * DIMC * NPX /*v*/, f3, DIMC, pp(22), pp(23), tdw, 2 * DIMC);
    gelu(tdw, th, (size_t)2 * DIMC * NPX);
    gemm(akv_v_pw_wh, th, pp(25), nullptr, v2b, DIMC, 2 * DIMC);
    k_rownorm<<<DIMC, 256, 0, stream>>>(qkvdw /*raw q*/, N);
    k_rownorm<<<DIMC, 256, 0, stream>>>(qn /*k2*/, N);
    k_attmat<<<NHEAD * HDIM * HDIM, 256, 0, stream>>>(qkvdw, qn, pp(13), Pm, N);
    k_topk_softmax<<<3, 128, 0, stream>>>(Pm, kkptr, NHEAD * HDIM);
    k_attn_av<<<dim3(N / 128, HDIM / 16, NHEAD), 256, 0, stream>>>(Pm, v2b, attoh, N);
    gemm(akv_proj_wh, attoh, pp(12), nullptr, outkv, DIMC, DIMC);

    // ======================= fusion ======================================
    dw(outq, outkv, DIMC, pp(26), pp(27), tdw, 2 * DIMC);
    gelu(tdw, th, (size_t)2 * DIMC * NPX);
    gemm(fus_pw_wh, th, pp(29), x /*residual*/, X1, DIMC, 2 * DIMC);

    // ======================= FFN =========================================
    k_layernorm<<<N / 256, 256, 0, stream>>>(X1, pp(32), pp(33), yh, DIMC, N);
    gemm(ffn_in_wh, yh, pp(35), nullptr, G1 /*hg*/, 2 * HIDC, DIMC);
    dw(G1, nullptr, 2 * HIDC, pp(36), pp(37), hgdw, 2 * HIDC);
    k_gate<<<(unsigned)(((size_t)HIDC * NPX + 255) / 256), 256, 0, stream>>>(
        hgdw, gh, (size_t)HIDC * NPX, (size_t)HIDC * NPX);
    gemm(ffn_out_wh, gh, pp(39), X1 /*residual*/, outp, DIMC, HIDC);

    (void)in_sizes; (void)n_in; (void)out_size; (void)ws_size;
}
